// MultiHeadedAttention_13168369729550
// MI455X (gfx1250) — matrix operand order repaired and checked by reading, not compiled, not run on hardware
//
#include <hip/hip_runtime.h>

#define HEADS 16
#define DMODEL 1024
#define DPH 64
#define SEQ 1024
#define BATCH 4
#define MAXREL 32
#define NREL 65

typedef __attribute__((ext_vector_type(16))) __bf16 v16bf;
typedef __attribute__((ext_vector_type(8)))  float  v8f;
typedef __attribute__((ext_vector_type(4)))  unsigned int v4u;
typedef __attribute__((ext_vector_type(8)))  int v8i;
typedef __attribute__((ext_vector_type(4)))  int v4i;

#if defined(__HIP_DEVICE_COMPILE__) && __has_builtin(__builtin_amdgcn_tensor_load_to_lds)
#define HAVE_TDM 1
#else
#define HAVE_TDM 0
#endif

#if defined(__HIP_DEVICE_COMPILE__) && __has_builtin(__builtin_amdgcn_s_wait_tensorcnt)
#define TDM_WAIT() __builtin_amdgcn_s_wait_tensorcnt(0)
#else
#define TDM_WAIT() do {} while (0)
#endif

// Low 32 bits of a flat pointer into LDS are the LDS byte offset
// (aperture check uses addr[63:32]).
__device__ __forceinline__ unsigned lds_byte_off(const void* p) {
  return (unsigned)(unsigned long long)p;
}

// ---------------------------------------------------------------------------
// TDM 2D tile load: global (row-major, stride_elems) -> LDS (packed rows).
// Descriptor packing per CDNA5 ISA ch.8 (D# group0 / group1; groups 2,3 = 0).
// data_size code: 0=1B 1=2B 2=4B 3=8B.
// ---------------------------------------------------------------------------
__device__ __forceinline__ void tdm_load_2d(const void* gptr, unsigned lds_off,
                                            unsigned ds_code,
                                            unsigned tensor_w, unsigned tensor_h,
                                            unsigned stride_elems,
                                            unsigned tile_w, unsigned tile_h) {
#if HAVE_TDM
  unsigned long long ga = (unsigned long long)gptr;
  v4u g0;
  g0[0] = 1u;                                           // count=1, user-mode D#
  g0[1] = lds_off;                                      // lds_addr
  g0[2] = (unsigned)(ga & 0xffffffffull);               // global_addr[31:0]
  g0[3] = (unsigned)((ga >> 32) & 0x1ffffffull) | (2u << 30); // ga[56:32] | type=2
  v8i g1;
  g1[0] = (int)(ds_code << 16);                         // wg_mask=0, data_size
  g1[1] = (int)((tensor_w & 0xffffu) << 16);            // abar=0, tensor_dim0[15:0]
  g1[2] = (int)((tensor_w >> 16) | ((tensor_h & 0xffffu) << 16)); // td0 hi | td1 lo
  g1[3] = (int)((tensor_h >> 16) | (tile_w << 16));     // td1 hi | tile_dim0
  g1[4] = (int)(tile_h & 0xffffu);                      // tile_dim1, tile_dim2=0
  g1[5] = (int)stride_elems;                            // tensor_dim0_stride[31:0]
  g1[6] = 0;                                            // stride hi | dim1_stride lo
  g1[7] = 0;
  v4i z4 = {};
#if __clang_major__ >= 23
  v8i z8 = {};
  __builtin_amdgcn_tensor_load_to_lds(g0, g1, z4, z4, z8, 0);
#else
  __builtin_amdgcn_tensor_load_to_lds(g0, g1, z4, z4, 0);
#endif
#else
  (void)gptr; (void)lds_off; (void)ds_code; (void)tensor_w; (void)tensor_h;
  (void)stride_elems; (void)tile_w; (void)tile_h;
#endif
}

// ---------------------------------------------------------------------------
// GEMM: C[M x N] = A[M x K] @ W[K x N] + bias, then * scale.
// AT = float (raw inputs) or __bf16 (workspace activations).
// OUT_BHSD: write bf16 into [B][H][S][DPH]; else fp32 row-major.
// WG: 256 threads / 8 waves; tile 64x64; wave owns two 16x16 accumulators.
// Double-buffered LDS staging via TDM (wave 0 issues, TENSORcnt tracked),
// with cooperative-load fallback if the builtin is unavailable.
// ---------------------------------------------------------------------------
template<typename AT, int OUT_BHSD>
__global__ __launch_bounds__(256)
void gemm_bias_kernel(const AT* __restrict__ A, const float* __restrict__ W,
                      const float* __restrict__ bias,
                      float* __restrict__ outF, __bf16* __restrict__ outB,
                      float scale, int M, int N, int K) {
  __shared__ AT    sA[2][64][32];
  __shared__ float sW[2][32][64];

  const int tid  = threadIdx.x;
  const int lane = tid & 31;
  const int wv   = tid >> 5;
  const int rw   = wv >> 2;          // 0..1 : row-tile pair
  const int wc   = wv & 3;           // 0..3 : col tile
  const int n0   = blockIdx.x * 64;
  const int m0   = blockIdx.y * 64;
  const int hi   = lane >> 4;
  const int lo16 = lane & 15;
  const unsigned dsA = (sizeof(AT) == 2) ? 1u : 2u;

  const int KT = K / 32;
  v8f acc0 = {}, acc1 = {};

  // stage k-tile kt into buffer buf
  auto stage = [&](int buf, int kt) {
    const AT*    gA = A + (size_t)m0 * K + kt * 32;
    const float* gW = W + (size_t)(kt * 32) * N + n0;
    if (HAVE_TDM) {
      if (wv == 0) {
        tdm_load_2d(gA, lds_byte_off(&sA[buf][0][0]), dsA,
                    (unsigned)K, (unsigned)M, (unsigned)K, 32u, 64u);
        tdm_load_2d(gW, lds_byte_off(&sW[buf][0][0]), 2u,
                    (unsigned)N, (unsigned)K, (unsigned)N, 64u, 32u);
      }
    } else {
      #pragma unroll
      for (int i = 0; i < 8; ++i) {               // 64x32 A tile
        int lin = tid + i * 256;
        int r = lin >> 5, c = lin & 31;
        sA[buf][r][c] = gA[(size_t)r * K + c];
      }
      #pragma unroll
      for (int i = 0; i < 8; ++i) {               // 32x64 W tile
        int lin = tid + i * 256;
        int r = lin >> 6, c = lin & 63;
        sW[buf][r][c] = gW[(size_t)r * N + c];
      }
    }
  };

  stage(0, 0);

  for (int kt = 0; kt < KT; ++kt) {
    const int buf = kt & 1;
    if (HAVE_TDM) { if (wv == 0) TDM_WAIT(); }
    __syncthreads();                               // staged tile visible to WG
    if (kt + 1 < KT) stage(buf ^ 1, kt + 1);       // prefetch next stage

    // build fragments from LDS (convert to bf16) and accumulate
    v16bf a0 = {}, a1 = {}, bf = {};
    const int row0 = rw * 32 + lo16;               // row tiles rw*2, rw*2+1
    const int bcol = wc * 16 + lo16;
    #pragma unroll
    for (int v = 0; v < 8; ++v) {
      int kb = (v >> 2) * 16 + hi * 8 + (v & 3) * 2;   // A-frag K pair
      a0[2 * v]     = (__bf16)sA[buf][row0][kb];
      a0[2 * v + 1] = (__bf16)sA[buf][row0][kb + 1];
      a1[2 * v]     = (__bf16)sA[buf][row0 + 16][kb];
      a1[2 * v + 1] = (__bf16)sA[buf][row0 + 16][kb + 1];
      int kp = (v >> 2) * 16 + hi * 8 + (v & 3) * 2;   // B-frag K pair
      bf[2 * v]     = (__bf16)sW[buf][kp][bcol];
      bf[2 * v + 1] = (__bf16)sW[buf][kp + 1][bcol];
    }
    acc0 = __builtin_amdgcn_wmma_f32_16x16x32_bf16(false, a0, false, bf,
                                                   (short)0, acc0, false, false);
    acc1 = __builtin_amdgcn_wmma_f32_16x16x32_bf16(false, a1, false, bf,
                                                   (short)0, acc1, false, false);
    __syncthreads();                               // done reading before overwrite
  }

  // writeback both 16x16 tiles (C/D layout: row = v + (lane/16)*8, col = lane%16)
  #pragma unroll
  for (int t = 0; t < 2; ++t) {
    v8f& acc = t ? acc1 : acc0;
    #pragma unroll
    for (int v = 0; v < 8; ++v) {
      int row = m0 + rw * 32 + t * 16 + v + hi * 8;
      int col = n0 + wc * 16 + lo16;
      float val = (acc[v] + bias[col]) * scale;
      if (OUT_BHSD) {
        int b = row >> 10, s = row & 1023;
        int h = col >> 6,  d = col & 63;
        outB[(((size_t)(b * HEADS + h) * SEQ + s) << 6) + d] = (__bf16)val;
      } else {
        outF[(size_t)row * N + col] = val;
      }
    }
  }
}

// ---------------------------------------------------------------------------
// Attention: one workgroup (256 threads / 8 waves) per (b, h, 16-query tile).
// qrel -> WMMA scores -> rel-add + mask + exact softmax (fp32 in LDS) ->
// top_attn (h==0) -> relative-bucket sums -> WMMA ctx + rel-ctx -> bf16 ctx.
// ---------------------------------------------------------------------------
__global__ __launch_bounds__(256)
void attn_kernel(const __bf16* __restrict__ qh, const __bf16* __restrict__ kh,
                 const __bf16* __restrict__ vh, const unsigned char* __restrict__ mask,
                 const float* __restrict__ rel_emb,
                 __bf16* __restrict__ ctx, float* __restrict__ top_attn) {
  __shared__ float sS[16][SEQ];       // 64 KB: score / attn rows
  __shared__ float sQrel[16][NREL];   // q . rel_emb[r]
  __shared__ float sAr[16][NREL];     // attn bucketed by relative distance
  __shared__ float sRed[16][16];      // row reductions
  __shared__ float sCtx[8][16][16];   // per-wave ctx partials

  const int tid  = threadIdx.x;
  const int lane = tid & 31;
  const int wv   = tid >> 5;
  const int hi   = lane >> 4;
  const int lo16 = lane & 15;

  const int qt = blockIdx.x, h = blockIdx.y, b = blockIdx.z;
  const int bh = b * HEADS + h;
  const int q0 = qt * 16;

  const __bf16* Q  = qh + (size_t)bh * SEQ * DPH;
  const __bf16* Kp = kh + (size_t)bh * SEQ * DPH;
  const __bf16* Vp = vh + (size_t)bh * SEQ * DPH;

  // ---- phase 0: qrel[q][r] = sum_d q[q0+q][d] * rel_emb[r][d] -------------
  for (int item = tid; item < 16 * NREL; item += 256) {
    int q = item & 15;
    int r = item >> 4;             // 0..64
    const __bf16* qrow = Q + (size_t)(q0 + q) * DPH;
    const float*  er   = rel_emb + r * DPH;
    float s = 0.f;
    #pragma unroll 8
    for (int d = 0; d < DPH; ++d) s += (float)qrow[d] * er[d];
    sQrel[q][r] = s;
  }

  // ---- phase 1: scores = Qtile @ K^T via WMMA, 8 key blocks per wave ------
  {
    v16bf a0 = {}, a1 = {};
    const __bf16* qr = Q + (size_t)(q0 + lo16) * DPH;
    #pragma unroll
    for (int v = 0; v < 8; ++v) {
      int kb = (v >> 2) * 16 + hi * 8 + (v & 3) * 2;   // feature index
      a0[2 * v]     = qr[kb];       a0[2 * v + 1] = qr[kb + 1];
      a1[2 * v]     = qr[32 + kb];  a1[2 * v + 1] = qr[32 + kb + 1];
    }
    for (int j = 0; j < 8; ++j) {
      int kblk = (wv * 8 + j) * 16;                    // key block base
      const __bf16* kr = Kp + (size_t)(kblk + lo16) * DPH;  // this lane's key row
      v16bf b0 = {}, b1 = {};
      #pragma unroll
      for (int v = 0; v < 8; ++v) {
        int kp = (v >> 2) * 16 + hi * 8 + (v & 3) * 2; // feature pair
        b0[2 * v]     = kr[kp];       b0[2 * v + 1] = kr[kp + 1];
        b1[2 * v]     = kr[32 + kp];  b1[2 * v + 1] = kr[32 + kp + 1];
      }
      v8f acc = {};
      acc = __builtin_amdgcn_wmma_f32_16x16x32_bf16(false, a0, false, b0,
                                                    (short)0, acc, false, false);
      acc = __builtin_amdgcn_wmma_f32_16x16x32_bf16(false, a1, false, b1,
                                                    (short)0, acc, false, false);
      #pragma unroll
      for (int v = 0; v < 8; ++v)
        sS[v + hi * 8][kblk + lo16] = acc[v];
    }
  }
  __syncthreads();

  // ---- phase 2: rel add + mask + exact softmax over 1024 keys -------------
  {
    const int q   = tid >> 4;      // row 0..15
    const int sub = tid & 15;
    const int qg  = q0 + q;
    float lmax = -3.0e38f;
    for (int i = 0; i < 64; ++i) {
      int k = sub + (i << 4);
      float s = sS[q][k];
      int dd = k - qg;
      dd = dd < -MAXREL ? -MAXREL : (dd > MAXREL ? MAXREL : dd);
      s += sQrel[q][dd + MAXREL];
      if (mask[b * SEQ + k]) s = -1e18f;
      sS[q][k] = s;
      lmax = fmaxf(lmax, s);
    }
    sRed[q][sub] = lmax;
    __syncthreads();
    float m = sRed[q][0];
    #pragma unroll
    for (int i = 1; i < 16; ++i) m = fmaxf(m, sRed[q][i]);
    __syncthreads();
    float lsum = 0.f;
    for (int i = 0; i < 64; ++i) {
      int k = sub + (i << 4);
      float e = __expf(sS[q][k] - m);
      sS[q][k] = e;
      lsum += e;
    }
    sRed[q][sub] = lsum;
    __syncthreads();
    float sum = 0.f;
    #pragma unroll
    for (int i = 0; i < 16; ++i) sum += sRed[q][i];
    float inv = 1.0f / sum;
    for (int i = 0; i < 64; ++i) sS[q][sub + (i << 4)] *= inv;
  }
  __syncthreads();

  // ---- phase 3a: emit head-0 attention (top_attn) -------------------------
  if (h == 0) {
    for (int j = 0; j < 64; ++j) {
      int lin = tid + j * 256;
      int q = lin >> 10, k = lin & 1023;
      top_attn[((size_t)(b * SEQ + q0 + q) << 10) + k] = sS[q][k];
    }
  }

  // ---- phase 3b: bucket attn by relative distance -------------------------
  for (int item = tid; item < 16 * NREL; item += 256) {
    int q = item & 15;
    int r = item >> 4;
    int qg = q0 + q;
    float s = 0.f;
    if (r == 0) {
      int kmax = qg - MAXREL;                 // k <= qg-32
      for (int k = 0; k <= kmax; ++k) s += sS[q][k];
    } else if (r == NREL - 1) {
      for (int k = qg + MAXREL; k < SEQ; ++k) s += sS[q][k];
    } else {
      int k = qg + r - MAXREL;
      s = (k >= 0 && k < SEQ) ? sS[q][k] : 0.f;
    }
    sAr[q][r] = s;
  }

  // ---- phase 4: ctx = attn @ V via WMMA (split keys across wave pairs) ----
  {
    int c  = wv & 3;               // d-column tile
    int hh = wv >> 2;              // key half
    int dcol = c * 16 + lo16;
    v8f acc = {};
    for (int t = 0; t < 16; ++t) {
      int kb0 = hh * 512 + t * 32;
      v16bf a = {}, bf = {};
      #pragma unroll
      for (int v = 0; v < 8; ++v) {
        int koff = (v >> 2) * 16 + hi * 8 + (v & 3) * 2;
        a[2 * v]     = (__bf16)sS[lo16][kb0 + koff];
        a[2 * v + 1] = (__bf16)sS[lo16][kb0 + koff + 1];
        int kk = kb0 + (v >> 2) * 16 + hi * 8 + (v & 3) * 2;
        bf[2 * v]     = Vp[(size_t)kk * DPH + dcol];
        bf[2 * v + 1] = Vp[(size_t)(kk + 1) * DPH + dcol];
      }
      acc = __builtin_amdgcn_wmma_f32_16x16x32_bf16(false, a, false, bf,
                                                    (short)0, acc, false, false);
    }
    #pragma unroll
    for (int v = 0; v < 8; ++v)
      sCtx[wv][v + hi * 8][lo16] = acc[v];
  }
  __syncthreads();

  // ---- phase 5: ctx += attn_r @ rel_emb, write bf16 [B][S][DMODEL] --------
  for (int item = tid; item < 16 * DPH; item += 256) {
    int q = item >> 6, d = item & 63;
    float s = 0.f;
    #pragma unroll 5
    for (int r = 0; r < NREL; ++r) s += sAr[q][r] * rel_emb[r * DPH + d];
    int c = d >> 4, n = d & 15;
    float val = sCtx[c][q][n] + sCtx[c + 4][q][n] + s;
    ctx[((size_t)(b * SEQ + q0 + q) << 10) + h * DPH + d] = (__bf16)val;
  }
}

// ---------------------------------------------------------------------------
extern "C" void kernel_launch(void* const* d_in, const int* in_sizes, int n_in,
                              void* d_out, int out_size, void* d_ws, size_t ws_size,
                              hipStream_t stream) {
  (void)in_sizes; (void)n_in; (void)out_size; (void)ws_size;

  const float* key   = (const float*)d_in[0];
  const float* value = (const float*)d_in[1];
  const float* query = (const float*)d_in[2];
  const unsigned char* mask = (const unsigned char*)d_in[3];
  const float* Wq = (const float*)d_in[4];
  const float* bq = (const float*)d_in[5];
  const float* Wk = (const float*)d_in[6];
  const float* bk = (const float*)d_in[7];
  const float* Wv = (const float*)d_in[8];
  const float* bv = (const float*)d_in[9];
  const float* Wo = (const float*)d_in[10];
  const float* bo = (const float*)d_in[11];
  const float* rel = (const float*)d_in[12];

  float* out = (float*)d_out;
  float* top = out + (size_t)BATCH * SEQ * DMODEL;

  char* ws = (char*)d_ws;
  __bf16* qh  = (__bf16*)(ws);                          //  8 MB
  __bf16* kh  = (__bf16*)(ws + (8ull  << 20));          //  8 MB
  __bf16* vh  = (__bf16*)(ws + (16ull << 20));          //  8 MB
  __bf16* ctx = (__bf16*)(ws + (24ull << 20));          //  8 MB

  const int M = BATCH * SEQ;      // 4096
  dim3 gblk(256);
  dim3 ggrid(DMODEL / 64, M / 64);

  // Q/K/V projections (Q pre-scaled by 1/sqrt(DPH) = 0.125)
  gemm_bias_kernel<float, 1><<<ggrid, gblk, 0, stream>>>(
      query, Wq, bq, nullptr, qh, 0.125f, M, DMODEL, DMODEL);
  gemm_bias_kernel<float, 1><<<ggrid, gblk, 0, stream>>>(
      key,   Wk, bk, nullptr, kh, 1.0f,   M, DMODEL, DMODEL);
  gemm_bias_kernel<float, 1><<<ggrid, gblk, 0, stream>>>(
      value, Wv, bv, nullptr, vh, 1.0f,   M, DMODEL, DMODEL);

  // attention
  dim3 agrid(SEQ / 16, HEADS, BATCH);
  attn_kernel<<<agrid, gblk, 0, stream>>>(qh, kh, vh, mask, rel, ctx, top);

  // output projection
  gemm_bias_kernel<__bf16, 0><<<ggrid, gblk, 0, stream>>>(
      ctx, Wo, bo, out, (__bf16*)nullptr, 1.0f, M, DMODEL, DMODEL);
}
